// PointNetChebDconv_61546881352049
// MI455X (gfx1250) — compile-verified, matrix operand order checked by reading, not run on hardware
//
#include <hip/hip_runtime.h>
#include <math.h>

// ---------------------------------------------------------------------------
// CDNA5 WMMA helpers (gfx1250, wave32)
// ---------------------------------------------------------------------------
typedef __attribute__((ext_vector_type(16))) _Float16 v16h;
typedef __attribute__((ext_vector_type(8)))  _Float16 v8h;
typedef __attribute__((ext_vector_type(8)))  float    v8f;

// 16x32 f16 A/B fragment layout (ISA 7.12.2): for lane-half hi, elements 0..7
// hold K = hi*8 .. hi*8+7 and elements 8..15 hold K = 16+hi*8 .. 23+hi*8.
// With K-contiguous 16B-aligned storage this is two b128 loads.
__device__ __forceinline__ v16h frag_ld(const _Float16* __restrict__ base, int hi) {
  v8h lo = *(const v8h*)(base + hi * 8);
  v8h hn = *(const v8h*)(base + 16 + hi * 8);
  v16h r;
#pragma unroll
  for (int i = 0; i < 8; ++i) { r[i] = lo[i]; r[8 + i] = hn[i]; }
  return r;
}

__device__ __forceinline__ void gemm_store(const v8f& c, float* __restrict__ C,
                                           const float* __restrict__ bias,
                                           const float* __restrict__ res,
                                           int acc, int relu,
                                           int row0, int col, int M, int N) {
#pragma unroll
  for (int r = 0; r < 8; ++r) {
    int row = row0 + r;
    if (row < M && col < N) {
      size_t o = (size_t)row * N + col;
      float v = c[r];
      if (acc)  v += C[o];
      if (bias) v += bias[col];
      if (res)  v += res[o];
      if (relu) v = fmaxf(v, 0.0f);
      C[o] = v;
    }
  }
}

// ---------------------------------------------------------------------------
// Generic GEMM: C[M,N] = act(A[M,K] * W[K,N] + bias (+C if acc) (+residual))
// A: f32 global, staged to zero-padded f16 LDS tile (segment-level fast path).
// W: pre-transposed zero-padded f16 Wt[Npad][Kp] (K-contiguous).
// 64x64 C tile per 128-thread block; 4 waves x (2x2) 16x16 subtiles:
// 2 A-frags + 2 B-frags -> 4 WMMAs per K-step.
// ---------------------------------------------------------------------------
__global__ void gemm_f16w_kernel(const float* __restrict__ A,
                                 const _Float16* __restrict__ Wt,
                                 const float* __restrict__ bias,
                                 const float* __restrict__ res,
                                 float* __restrict__ C,
                                 int M, int K, int Kp, int N, int relu, int acc) {
  __shared__ _Float16 As[64 * 32];
  const int tid  = threadIdx.x;
  const int lane = tid & 31;
  const int wid  = tid >> 5;          // 0..3
  const int tilesN = (N + 63) >> 6;
  const int tm64 = (blockIdx.x / tilesN) << 6;
  const int tn64 = (blockIdx.x % tilesN) << 6;
  const int hi  = lane >> 4;
  const int l16 = lane & 15;
  const int smA = (wid >> 1) << 5;    // 0 or 32: wave's row-quadrant
  const int snA = (wid & 1) << 5;     // 0 or 32: wave's col-quadrant
  const int col0 = tn64 + snA + l16;
  const int col1 = col0 + 16;
  const _Float16* bptr0 = Wt + (size_t)col0 * Kp;  // always valid (Npad pad)
  const _Float16* bptr1 = Wt + (size_t)col1 * Kp;
  // staging assignment: thread covers one contiguous 16-half segment
  const int srow = tid >> 1;          // 0..63
  const int skk  = (tid & 1) << 4;    // 0 or 16
  v8f c00 = {}, c01 = {}, c10 = {}, c11 = {};
  for (int k0 = 0; k0 < Kp; k0 += 32) {
    __syncthreads();
    {
      int gr = tm64 + srow;
      _Float16 tmp[16];
      if (gr < M && (k0 + skk + 15) < K) {
        const float* src = A + (size_t)gr * K + k0 + skk;
#pragma unroll
        for (int i = 0; i < 16; ++i) tmp[i] = (_Float16)src[i];
      } else {
#pragma unroll
        for (int i = 0; i < 16; ++i) {
          int gk = k0 + skk + i;
          tmp[i] = (gr < M && gk < K) ? (_Float16)A[(size_t)gr * K + gk] : (_Float16)0.0f;
        }
      }
      v8h s0, s1;
#pragma unroll
      for (int i = 0; i < 8; ++i) { s0[i] = tmp[i]; s1[i] = tmp[8 + i]; }
      *(v8h*)&As[srow * 32 + skk]     = s0;
      *(v8h*)&As[srow * 32 + skk + 8] = s1;
    }
    __syncthreads();
    v16h b0 = frag_ld(bptr0 + k0, hi);
    v16h b1 = frag_ld(bptr1 + k0, hi);
    v16h a0 = frag_ld(As + (smA + l16) * 32, hi);
    v16h a1 = frag_ld(As + (smA + 16 + l16) * 32, hi);
    c00 = __builtin_amdgcn_wmma_f32_16x16x32_f16(false, a0, false, b0, (short)0, c00, false, false);
    c01 = __builtin_amdgcn_wmma_f32_16x16x32_f16(false, a0, false, b1, (short)0, c01, false, false);
    c10 = __builtin_amdgcn_wmma_f32_16x16x32_f16(false, a1, false, b0, (short)0, c10, false, false);
    c11 = __builtin_amdgcn_wmma_f32_16x16x32_f16(false, a1, false, b1, (short)0, c11, false, false);
  }
  gemm_store(c00, C, bias, res, acc, relu, tm64 + smA + hi * 8,      col0, M, N);
  gemm_store(c01, C, bias, res, acc, relu, tm64 + smA + hi * 8,      col1, M, N);
  gemm_store(c10, C, bias, res, acc, relu, tm64 + smA + 16 + hi * 8, col0, M, N);
  gemm_store(c11, C, bias, res, acc, relu, tm64 + smA + 16 + hi * 8, col1, M, N);
}

// ---------------------------------------------------------------------------
// In-LDS MLP layer for the fused PointNetConv kernel. A (f16 LDS) is
// MROWS x Kp (Kp multiple of 32, zero-padded). Wt is Npad x Kp (global f16).
// N, MROWS multiples of 16. Branch-free inner loop.
// ---------------------------------------------------------------------------
template <int MROWS>
__device__ __forceinline__ void mlp_layer(const _Float16* __restrict__ A, int Kp,
                                          const _Float16* __restrict__ Wt, int N,
                                          const float* __restrict__ bias, bool relu,
                                          _Float16* out16, float* out32) {
  const int lane = threadIdx.x & 31;
  const int wid  = threadIdx.x >> 5;
  const int nw   = blockDim.x >> 5;
  const int tilesN = N >> 4;
  const int tilesM = MROWS >> 4;
  const int hi  = lane >> 4;
  const int l16 = lane & 15;
  for (int t = wid; t < tilesM * tilesN; t += nw) {
    int tm = (t / tilesN) << 4;
    int tn = (t % tilesN) << 4;
    const _Float16* arow = A  + (size_t)(tm + l16) * Kp;
    const _Float16* brow = Wt + (size_t)(tn + l16) * Kp;
    v8f c = {};
    for (int k0 = 0; k0 < Kp; k0 += 32) {
      v16h a = frag_ld(arow + k0, hi);
      v16h b = frag_ld(brow + k0, hi);
      c = __builtin_amdgcn_wmma_f32_16x16x32_f16(false, a, false, b, (short)0, c, false, false);
    }
#pragma unroll
    for (int r = 0; r < 8; ++r) {
      int row = tm + r + hi * 8;
      int colc = tn + l16;
      float v = c[r] + (bias ? bias[colc] : 0.0f);
      if (relu) v = fmaxf(v, 0.0f);
      if (out16) out16[row * N + colc] = (_Float16)v;
      if (out32) out32[row * N + colc] = v;
    }
  }
}

// ---------------------------------------------------------------------------
// Fused PointNetConv: one block per query; gather KN neighbor features
// [x_j, pos_j - q] into zero-padded f16 LDS (stride INP), 3-layer WMMA MLP
// entirely in LDS, then radius-masked max.
// ---------------------------------------------------------------------------
template <int INP, int H1, int H2, int OUT, int KN>
constexpr size_t pnconv_smem() {
  size_t o = (size_t)KN * INP * 2;
  o = (o + 15) & ~(size_t)15; o += (size_t)KN * H1 * 2;
  o = (o + 15) & ~(size_t)15; o += (size_t)KN * H2 * 2;
  o = (o + 15) & ~(size_t)15; o += (size_t)KN * OUT * 4;
  o += (size_t)KN * 4;  // nbr
  o += (size_t)KN * 4;  // d2
  return o;
}

template <int INP, int H1, int H2, int OUT, int KN>
__global__ void pnconv_kernel(const float* __restrict__ X, int xdim,
                              const float* __restrict__ P,  // n x 3 (deformed pos)
                              const float* __restrict__ Q,  // m x 3 (queries)
                              const int* __restrict__ NBR, const float* __restrict__ D2,
                              const _Float16* __restrict__ W0, const float* __restrict__ B0,
                              const _Float16* __restrict__ W1, const float* __restrict__ B1,
                              const _Float16* __restrict__ W2, const float* __restrict__ B2,
                              float r2, float* __restrict__ Y) {
  extern __shared__ char smem[];
  constexpr size_t OF_H1  = (((size_t)KN * INP * 2) + 15) & ~(size_t)15;
  constexpr size_t OF_H2  = ((OF_H1 + (size_t)KN * H1 * 2) + 15) & ~(size_t)15;
  constexpr size_t OF_H3  = ((OF_H2 + (size_t)KN * H2 * 2) + 15) & ~(size_t)15;
  constexpr size_t OF_NBR = OF_H3 + (size_t)KN * OUT * 4;
  constexpr size_t OF_D2  = OF_NBR + (size_t)KN * 4;
  _Float16* feats = (_Float16*)smem;
  _Float16* h1    = (_Float16*)(smem + OF_H1);
  _Float16* h2    = (_Float16*)(smem + OF_H2);
  float*    h3    = (float*)(smem + OF_H3);
  int*      nbr_s = (int*)(smem + OF_NBR);
  float*    d2_s  = (float*)(smem + OF_D2);

  const int q   = blockIdx.x;
  const int tid = threadIdx.x;
  for (int j = tid; j < KN; j += blockDim.x) {
    nbr_s[j] = NBR[q * KN + j];
    d2_s[j]  = D2[q * KN + j];
  }
  __syncthreads();
  const int in_dim = xdim + 3;
  for (int idx = tid; idx < KN * INP; idx += blockDim.x) {
    int j = idx / INP, c = idx % INP;
    int pj = nbr_s[j];
    _Float16 v = (_Float16)0.0f;
    if (c < xdim)       v = (_Float16)X[(size_t)pj * xdim + c];
    else if (c < in_dim) { int d = c - xdim; v = (_Float16)(P[pj * 3 + d] - Q[q * 3 + d]); }
    feats[idx] = v;
  }
  __syncthreads();
  mlp_layer<KN>(feats, INP, W0, H1, B0, true, h1, nullptr);
  __syncthreads();
  mlp_layer<KN>(h1, H1, W1, H2, B1, true, h2, nullptr);
  __syncthreads();
  mlp_layer<KN>(h2, H2, W2, OUT, B2, false, nullptr, h3);
  __syncthreads();
  for (int c = tid; c < OUT; c += blockDim.x) {
    float mx = -INFINITY;
    for (int j = 0; j < KN; ++j)
      if (d2_s[j] <= r2) mx = fmaxf(mx, h3[j * OUT + c]);
    Y[(size_t)q * OUT + c] = mx;
  }
}

// ---------------------------------------------------------------------------
// Farthest point sampling (one block), matches reference scan semantics.
// ---------------------------------------------------------------------------
__global__ void fps_kernel(const float* __restrict__ P, int n, int m, int* __restrict__ idx) {
  extern __shared__ char smem[];
  float* dbuf = (float*)smem;
  float* rv   = dbuf + n;
  int*   ri   = (int*)(rv + blockDim.x);
  float* cur  = (float*)(ri + blockDim.x);
  const int tid = threadIdx.x, nt = blockDim.x;
  if (tid == 0) { idx[0] = 0; cur[0] = P[0]; cur[1] = P[1]; cur[2] = P[2]; }
  __syncthreads();
  for (int j = tid; j < n; j += nt) {
    float dx = P[j*3]-cur[0], dy = P[j*3+1]-cur[1], dz = P[j*3+2]-cur[2];
    dbuf[j] = dx*dx + dy*dy + dz*dz;
  }
  __syncthreads();
  for (int t = 1; t < m; ++t) {
    float bv = -INFINITY; int bi = 0x7fffffff;
    for (int j = tid; j < n; j += nt) {
      float v = dbuf[j];
      if (v > bv || (v == bv && j < bi)) { bv = v; bi = j; }
    }
    rv[tid] = bv; ri[tid] = bi;
    __syncthreads();
    for (int s = nt >> 1; s > 0; s >>= 1) {
      if (tid < s) {
        if (rv[tid+s] > rv[tid] || (rv[tid+s] == rv[tid] && ri[tid+s] < ri[tid])) {
          rv[tid] = rv[tid+s]; ri[tid] = ri[tid+s];
        }
      }
      __syncthreads();
    }
    if (tid == 0) {
      int w = ri[0];
      idx[t] = w;
      cur[0] = P[w*3]; cur[1] = P[w*3+1]; cur[2] = P[w*3+2];
    }
    __syncthreads();
    for (int j = tid; j < n; j += nt) {
      float dx = P[j*3]-cur[0], dy = P[j*3+1]-cur[1], dz = P[j*3+2]-cur[2];
      dbuf[j] = fminf(dbuf[j], dx*dx + dy*dy + dz*dz);
    }
    __syncthreads();
  }
}

// ---------------------------------------------------------------------------
// kNN: one block per query; LDS-resident d2; k x argmin-extract (ascending).
// ---------------------------------------------------------------------------
__global__ void knn_kernel(const float* __restrict__ Q, const float* __restrict__ P,
                           int n, int k, int* __restrict__ nbr, float* __restrict__ d2out) {
  extern __shared__ char smem[];
  float* dbuf = (float*)smem;
  float* rv   = dbuf + n;
  int*   ri   = (int*)(rv + blockDim.x);
  const int q = blockIdx.x, tid = threadIdx.x, nt = blockDim.x;
  const float qx = Q[q*3], qy = Q[q*3+1], qz = Q[q*3+2];
  for (int j = tid; j < n; j += nt) {
    float dx = P[j*3]-qx, dy = P[j*3+1]-qy, dz = P[j*3+2]-qz;
    dbuf[j] = dx*dx + dy*dy + dz*dz;
  }
  __syncthreads();
  for (int t = 0; t < k; ++t) {
    float bv = INFINITY; int bi = 0x7fffffff;
    for (int j = tid; j < n; j += nt) {
      float v = dbuf[j];
      if (v < bv || (v == bv && j < bi)) { bv = v; bi = j; }
    }
    rv[tid] = bv; ri[tid] = bi;
    __syncthreads();
    for (int s = nt >> 1; s > 0; s >>= 1) {
      if (tid < s) {
        if (rv[tid+s] < rv[tid] || (rv[tid+s] == rv[tid] && ri[tid+s] < ri[tid])) {
          rv[tid] = rv[tid+s]; ri[tid] = ri[tid+s];
        }
      }
      __syncthreads();
    }
    if (tid == 0) {
      int w = ri[0];
      nbr[q*k + t] = w; d2out[q*k + t] = rv[0];
      dbuf[w] = INFINITY;
    }
    __syncthreads();
  }
}

// ---------------------------------------------------------------------------
// Small elementwise / gather / scatter kernels
// ---------------------------------------------------------------------------
__global__ void convert_weight_kernel(const float* __restrict__ W, _Float16* __restrict__ Wt,
                                      int K, int N, int Kp, int Np) {
  int t = blockIdx.x * blockDim.x + threadIdx.x;
  if (t < Np * Kp) {
    int n = t / Kp, k = t % Kp;
    Wt[t] = (n < N && k < K) ? (_Float16)W[(size_t)k * N + n] : (_Float16)0.0f;
  }
}
__global__ void zero_kernel(float* p, int n) {
  int t = blockIdx.x * blockDim.x + threadIdx.x;
  if (t < n) p[t] = 0.0f;
}
__global__ void axpb_kernel(const float* __restrict__ S, const float* __restrict__ T0,
                            float* __restrict__ out, float alpha, float beta, int n) {
  int t = blockIdx.x * blockDim.x + threadIdx.x;
  if (t < n) out[t] = alpha * S[t] + (T0 ? beta * T0[t] : 0.0f);
}
__global__ void gather_rows_kernel(const float* __restrict__ src, const int* __restrict__ idx,
                                   float* __restrict__ dst, int m, int C) {
  int t = blockIdx.x * blockDim.x + threadIdx.x;
  if (t < m * C) { int i = t / C, c = t % C; dst[t] = src[(size_t)idx[i] * C + c]; }
}
__global__ void gather_max_kernel(const float* __restrict__ H, const int* __restrict__ nbr,
                                  float* __restrict__ out, int m, int k, int C) {
  int t = blockIdx.x * blockDim.x + threadIdx.x;
  if (t < m * C) {
    int i = t / C, c = t % C;
    float mx = -INFINITY;
    for (int j = 0; j < k; ++j) mx = fmaxf(mx, H[(size_t)nbr[i*k+j] * C + c]);
    out[t] = mx;
  }
}
__global__ void scatter_add_kernel(const float* __restrict__ V, const int* __restrict__ nbr,
                                   float* __restrict__ out, int n, int k, int C) {
  int t = blockIdx.x * blockDim.x + threadIdx.x;
  if (t < n * C) {
    int i = t / C, c = t % C;
    float v = V[t];
    for (int j = 0; j < k; ++j) atomicAdd(&out[(size_t)nbr[i*k+j] * C + c], v);
  }
}
__global__ void concat_kernel(const float* __restrict__ A, int Ca,
                              const float* __restrict__ B, int Cb,
                              float* __restrict__ out, int m) {
  int t = blockIdx.x * blockDim.x + threadIdx.x;
  int C = Ca + Cb;
  if (t < m * C) {
    int i = t / C, c = t % C;
    out[t] = (c < Ca) ? A[(size_t)i * Ca + c] : B[(size_t)i * Cb + (c - Ca)];
  }
}
__global__ void mean_rows_kernel(const float* __restrict__ X, int m, int C, float* __restrict__ out) {
  int c = blockIdx.x * blockDim.x + threadIdx.x;
  if (c < C) {
    float s = 0.0f;
    for (int i = 0; i < m; ++i) s += X[(size_t)i * C + c];
    out[c] = s / (float)m;
  }
}

// ---------------------------------------------------------------------------
// Host orchestration
// ---------------------------------------------------------------------------
static inline dim3 elw(int n) { return dim3((n + 255) / 256); }

extern "C" void kernel_launch(void* const* d_in, const int* in_sizes, int n_in,
                              void* d_out, int out_size, void* d_ws, size_t ws_size,
                              hipStream_t stream) {
  (void)in_sizes; (void)n_in; (void)out_size; (void)ws_size;
  auto F = [&](int i) { return (const float*)d_in[i]; };

  char* ws = (char*)d_ws;
  size_t off = 0;
  auto alloc = [&](size_t bytes) -> void* {
    size_t a = (off + 255) & ~(size_t)255;
    off = a + bytes;
    return (void*)(ws + a);
  };

  auto gemm = [&](const float* A, const _Float16* Wt, const float* bias, const float* res,
                  float* C, int M, int K, int Kp, int N, int relu, int acc) {
    int blocks = ((M + 63) / 64) * ((N + 63) / 64);
    gemm_f16w_kernel<<<blocks, 128, 0, stream>>>(A, Wt, bias, res, C, M, K, Kp, N, relu, acc);
  };

  // ---- Convert weights to transposed, zero-padded f16 Wt[Npad][Kp] ----
  struct WSpec { int idx; int srcoff; int K; int N; };
  const WSpec wspec[32] = {
    {1,0,3,32},   {3,0,32,64},  {5,0,64,3},      // deform1
    {7,0,3,32},   {9,0,32,64},  {11,0,64,3},     // deform_td1
    {13,0,3,32},  {15,0,32,64}, {17,0,64,3},     // deform2
    {19,0,3,32},  {21,0,32,64}, {23,0,64,3},     // deform_td2
    {25,0,6,64},  {27,0,64,64}, {29,0,64,128},   // sa1_nn
    {31,0,128,256},                              // td1
    {33,0,256,256},{33,65536,256,256},{33,131072,256,256},  // cheb1 (3 mats)
    {35,0,259,128},{37,0,128,128},{39,0,128,256},// sa2_nn
    {41,0,256,512},                              // td2
    {43,0,512,512},{43,262144,512,512},{43,524288,512,512}, // cheb2 (3 mats)
    {45,0,515,256},{47,0,256,512},{49,0,512,1024},          // global_nn
    {51,0,1024,512},{53,0,512,256},{55,0,256,40}            // head
  };
  _Float16* w16[32];
  int wKp[32];
  for (int i = 0; i < 32; ++i) {
    int K = wspec[i].K, N = wspec[i].N;
    int Kp = (K + 31) & ~31;
    int Np = (N + 63) & ~63;
    wKp[i] = Kp;
    w16[i] = (_Float16*)alloc((size_t)Np * Kp * 2);
    convert_weight_kernel<<<elw(Np * Kp), 256, 0, stream>>>(
        F(wspec[i].idx) + wspec[i].srcoff, w16[i], K, N, Kp, Np);
  }

  // ---- Scratch buffers (reused across the 8-sample loop) ----
  float* dpos1  = (float*)alloc(4096*3*4);
  float* tA     = (float*)alloc(4096*64*4);
  float* tB     = (float*)alloc(4096*64*4);
  int*   idx1   = (int*)alloc(2048*4);
  float* pos1   = (float*)alloc(2048*3*4);
  int*   nbr1   = (int*)alloc(2048*64*4);
  float* d21    = (float*)alloc(2048*64*4);
  float* x1     = (float*)alloc(2048*128*4);
  float* dposT1 = (float*)alloc(2048*3*4);
  int*   idxT1  = (int*)alloc(512*4);
  float* pos2   = (float*)alloc(512*3*4);
  int*   nbrT1  = (int*)alloc(512*16*4);
  float* d2T1   = (float*)alloc(512*16*4);
  float* hT1    = (float*)alloc(2048*256*4);
  float* x2     = (float*)alloc(512*256*4);
  int*   nbrC1  = (int*)alloc(512*16*4);
  float* d2C1   = (float*)alloc(512*16*4);
  float* sbuf   = (float*)alloc(512*256*4);
  float* t1     = (float*)alloc(512*256*4);
  float* t2     = (float*)alloc(512*256*4);
  float* x2b    = (float*)alloc(512*256*4);
  float* dpos3  = (float*)alloc(512*3*4);
  int*   idx2   = (int*)alloc(128*4);
  float* pos3   = (float*)alloc(128*3*4);
  int*   nbr2   = (int*)alloc(128*64*4);
  float* d22    = (float*)alloc(128*64*4);
  float* x3     = (float*)alloc(128*256*4);
  float* dposT2 = (float*)alloc(128*3*4);
  int*   idxT2  = (int*)alloc(32*4);
  float* pos4   = (float*)alloc(32*3*4);
  int*   nbrT2  = (int*)alloc(32*16*4);
  float* d2T2   = (float*)alloc(32*16*4);
  float* hT2    = (float*)alloc(128*512*4);
  float* x4     = (float*)alloc(32*512*4);
  int*   nbrC2  = (int*)alloc(32*16*4);
  float* d2C2   = (float*)alloc(32*16*4);
  float* s2b    = (float*)alloc(32*512*4);
  float* t1b    = (float*)alloc(32*512*4);
  float* t2b    = (float*)alloc(32*512*4);
  float* x4b    = (float*)alloc(32*512*4);
  float* cat    = (float*)alloc(32*515*4);
  float* g1     = (float*)alloc(32*256*4);
  float* g2     = (float*)alloc(32*512*4);
  float* g3     = (float*)alloc(32*1024*4);
  float* featAll= (float*)alloc(8*1024*4);
  float* hh1    = (float*)alloc(8*512*4);
  float* hh2    = (float*)alloc(8*256*4);

  // Allow large dynamic LDS for the fused convs (SA2 ~134 KB; CDNA5 WG max 320 KB).
  (void)hipFuncSetAttribute(
      reinterpret_cast<const void*>(&pnconv_kernel<32,64,64,128,64>),
      hipFuncAttributeMaxDynamicSharedMemorySize, (int)pnconv_smem<32,64,64,128,64>());
  (void)hipFuncSetAttribute(
      reinterpret_cast<const void*>(&pnconv_kernel<288,128,128,256,64>),
      hipFuncAttributeMaxDynamicSharedMemorySize, (int)pnconv_smem<288,128,128,256,64>());

  auto fps = [&](const float* P, int n, int m, int* idx) {
    size_t sh = (size_t)n*4 + 256*4 + 256*4 + 4*4;
    fps_kernel<<<1, 256, sh, stream>>>(P, n, m, idx);
  };
  auto knn = [&](const float* Q, const float* P, int mq, int n, int k, int* nbr, float* d2) {
    size_t sh = (size_t)n*4 + 256*4 + 256*4;
    knn_kernel<<<mq, 256, sh, stream>>>(Q, P, n, k, nbr, d2);
  };
  auto deform = [&](const float* pos, int n, int w0, const float* b0,
                    int w1, const float* b1, int w2, const float* b2, float* out) {
    gemm(pos, w16[w0], b0, nullptr, tA, n, 3,  wKp[w0], 32, 1, 0);
    gemm(tA,  w16[w1], b1, nullptr, tB, n, 32, wKp[w1], 64, 1, 0);
    gemm(tB,  w16[w2], b2, pos,     out, n, 64, wKp[w2], 3, 0, 0);
  };
  auto cheb = [&](const float* x, int n, int C, const int* nbrn, int wbase,
                  const float* bc, float* s, float* t1p, float* t2p, float* out) {
    int cnt = n * C;
    zero_kernel<<<elw(cnt), 256, 0, stream>>>(s, cnt);
    scatter_add_kernel<<<elw(cnt), 256, 0, stream>>>(x, nbrn, s, n, 16, C);
    axpb_kernel<<<elw(cnt), 256, 0, stream>>>(s, nullptr, t1p, -1.0f/16.0f, 0.0f, cnt);
    zero_kernel<<<elw(cnt), 256, 0, stream>>>(s, cnt);
    scatter_add_kernel<<<elw(cnt), 256, 0, stream>>>(t1p, nbrn, s, n, 16, C);
    axpb_kernel<<<elw(cnt), 256, 0, stream>>>(s, x, t2p, -2.0f/16.0f, -1.0f, cnt);
    gemm(x,   w16[wbase],   bc,      nullptr, out, n, C, wKp[wbase],   C, 0, 0);
    gemm(t1p, w16[wbase+1], nullptr, nullptr, out, n, C, wKp[wbase+1], C, 0, 1);
    gemm(t2p, w16[wbase+2], nullptr, nullptr, out, n, C, wKp[wbase+2], C, 0, 1);
  };

  for (int b = 0; b < 8; ++b) {
    const float* pos0 = F(0) + (size_t)b * 4096 * 3;

    // --- SA1: deform1 -> fps(2048) -> knn64 -> fused PointNetConv [6,64,64,128]
    deform(pos0, 4096, 0, F(2), 1, F(4), 2, F(6), dpos1);
    fps(dpos1, 4096, 2048, idx1);
    gather_rows_kernel<<<elw(2048*3), 256, 0, stream>>>(dpos1, idx1, pos1, 2048, 3);
    knn(pos1, dpos1, 2048, 4096, 64, nbr1, d21);
    pnconv_kernel<32,64,64,128,64>
        <<<2048, 256, pnconv_smem<32,64,64,128,64>(), stream>>>(
        pos0, 3, dpos1, pos1, nbr1, d21,
        w16[12], F(26), w16[13], F(28), w16[14], F(30), 0.04f, x1);

    // --- TransitionDown1
    deform(pos1, 2048, 3, F(8), 4, F(10), 5, F(12), dposT1);
    fps(dposT1, 2048, 512, idxT1);
    gather_rows_kernel<<<elw(512*3), 256, 0, stream>>>(dposT1, idxT1, pos2, 512, 3);
    knn(pos2, dposT1, 512, 2048, 16, nbrT1, d2T1);
    gemm(x1, w16[15], F(32), nullptr, hT1, 2048, 128, wKp[15], 256, 1, 0);
    gather_max_kernel<<<elw(512*256), 256, 0, stream>>>(hT1, nbrT1, x2, 512, 16, 256);

    // --- ChebConv1 (knn16 graph on pos2)
    knn(pos2, pos2, 512, 512, 16, nbrC1, d2C1);
    cheb(x2, 512, 256, nbrC1, 16, F(34), sbuf, t1, t2, x2b);

    // --- SA2: deform2 -> fps(128) -> knn64 -> fused PointNetConv [259,128,128,256]
    deform(pos2, 512, 6, F(14), 7, F(16), 8, F(18), dpos3);
    fps(dpos3, 512, 128, idx2);
    gather_rows_kernel<<<elw(128*3), 256, 0, stream>>>(dpos3, idx2, pos3, 128, 3);
    knn(pos3, dpos3, 128, 512, 64, nbr2, d22);
    pnconv_kernel<288,128,128,256,64>
        <<<128, 256, pnconv_smem<288,128,128,256,64>(), stream>>>(
        x2b, 256, dpos3, pos3, nbr2, d22,
        w16[19], F(36), w16[20], F(38), w16[21], F(40), 0.16f, x3);

    // --- TransitionDown2
    deform(pos3, 128, 9, F(20), 10, F(22), 11, F(24), dposT2);
    fps(dposT2, 128, 32, idxT2);
    gather_rows_kernel<<<elw(32*3), 256, 0, stream>>>(dposT2, idxT2, pos4, 32, 3);
    knn(pos4, dposT2, 32, 128, 16, nbrT2, d2T2);
    gemm(x3, w16[22], F(42), nullptr, hT2, 128, 256, wKp[22], 512, 1, 0);
    gather_max_kernel<<<elw(32*512), 256, 0, stream>>>(hT2, nbrT2, x4, 32, 16, 512);

    // --- ChebConv2
    knn(pos4, pos4, 32, 32, 16, nbrC2, d2C2);
    cheb(x4, 32, 512, nbrC2, 23, F(44), s2b, t1b, t2b, x4b);

    // --- Global MLP + mean pool
    concat_kernel<<<elw(32*515), 256, 0, stream>>>(x4b, 512, pos4, 3, cat, 32);
    gemm(cat, w16[26], F(46), nullptr, g1, 32, 515, wKp[26], 256, 1, 0);
    gemm(g1,  w16[27], F(48), nullptr, g2, 32, 256, wKp[27], 512, 1, 0);
    gemm(g2,  w16[28], F(50), nullptr, g3, 32, 512, wKp[28], 1024, 0, 0);
    mean_rows_kernel<<<elw(1024), 256, 0, stream>>>(g3, 32, 1024, featAll + (size_t)b * 1024);
  }

  // --- Classifier head on [8,1024]
  gemm(featAll, w16[29], F(52), nullptr, hh1, 8, 1024, wKp[29], 512, 1, 0);
  gemm(hh1,     w16[30], F(54), nullptr, hh2, 8, 512,  wKp[30], 256, 1, 0);
  gemm(hh2,     w16[31], F(56), nullptr, (float*)d_out, 8, 256, wKp[31], 40, 0, 0);
}
